// MultiHeadDotProductAttention_54262616818022
// MI455X (gfx1250) — matrix operand order repaired and checked by reading, not compiled, not run on hardware
//
#include <hip/hip_runtime.h>
#include <hip/hip_bf16.h>

#define DEV __device__ __forceinline__

typedef __attribute__((ext_vector_type(16))) _Float16 v16h;
typedef __attribute__((ext_vector_type(8)))  _Float16 v8h;
typedef __attribute__((ext_vector_type(8)))  float    v8f;
typedef __attribute__((ext_vector_type(4)))  unsigned v4u;
typedef __attribute__((ext_vector_type(8)))  int      v8i;
typedef __attribute__((ext_vector_type(4)))  int      v4i;

// Problem constants: B=128, T=64, D=1024, H=16, HD=64
union F16Frag { v16h v; v8h h[2]; };

// A-matrix fragment (16x32, f16) from LDS stored row-major [M][K].
DEV v16h fragA_lds(const _Float16* A, int lda, int m0, int k0) {
  const int l = threadIdx.x & 31;
  const _Float16* p = A + (m0 + (l & 15)) * lda + k0 + ((l >> 4) << 3);
  F16Frag f;
  f.h[0] = *(const v8h*)(p);
  f.h[1] = *(const v8h*)(p + 16);
  return f.v;
}

// B-matrix fragment (32x16, f16) from LDS stored transposed [N][K].
DEV v16h fragB_lds(const _Float16* Bt, int ldb, int n0, int k0) {
  const int l = threadIdx.x & 31;
  const _Float16* p = Bt + (n0 + (l & 15)) * ldb + k0 + ((l >> 4) << 3);
  F16Frag f;
  f.h[0] = *(const v8h*)(p);
  f.h[1] = *(const v8h*)(p + 16);
  return f.v;
}

DEV v8f wmma_f16(v16h a, v16h b, v8f c) {
  return __builtin_amdgcn_wmma_f32_16x16x32_f16(false, a, false, b, (short)0, c,
                                                false, false);
}

// ---------------- Tensor Data Mover (TDM) descriptors ----------------
// 2D tile of f16 (data_size=1 => 2 bytes). Dims/strides in elements.
// D# group0: [1:0]=count, [63:32]=lds_addr, [120:64]=global_addr, [127:126]=type(2)
// D# group1: [17:16]=data_size, [79:48]=tensor_dim0, [111:80]=tensor_dim1,
//            [127:112]=tile_dim0, [143:128]=tile_dim1, [207:160]=tensor_dim0_stride
struct TdmDesc { v4u g0; v8i g1; };

DEV unsigned lds_u32(const void* p) {  // LDS byte offset = low 32b of generic addr
  return (unsigned)(unsigned long long)p;
}

DEV TdmDesc make_tdm_2d_f16(unsigned lds_byte, const void* gaddr,
                            unsigned tile_x, unsigned tile_y,
                            unsigned tensor_x, unsigned tensor_y,
                            unsigned long long stride0) {
  unsigned long long ga = (unsigned long long)gaddr;
  TdmDesc d;
  d.g0[0] = 1u;                                   // count=1, user D#
  d.g0[1] = lds_byte;
  d.g0[2] = (unsigned)ga;
  d.g0[3] = (unsigned)(ga >> 32) | (2u << 30);    // type=2 ("image")
  d.g1[0] = (int)(1u << 16);                      // data_size=1 (2B), mask=0
  d.g1[1] = (int)((tensor_x & 0xFFFFu) << 16);    // barrier_addr=0 | dim0.lo
  d.g1[2] = (int)((tensor_x >> 16) | ((tensor_y & 0xFFFFu) << 16));
  d.g1[3] = (int)((tensor_y >> 16) | (tile_x << 16));
  d.g1[4] = (int)tile_y;                          // tile_dim1 | tile_dim2=0
  d.g1[5] = (int)(unsigned)stride0;
  d.g1[6] = (int)((stride0 >> 32) & 0xFFFFu);     // stride1=0
  d.g1[7] = 0;
  return d;
}

DEV void tdm_load(const TdmDesc& d) {
  v4i z = {0, 0, 0, 0};
#if defined(__clang_major__) && __clang_major__ >= 23
  v8i z8 = {0, 0, 0, 0, 0, 0, 0, 0};
  __builtin_amdgcn_tensor_load_to_lds(d.g0, d.g1, z, z, z8, 0);
#else
  __builtin_amdgcn_tensor_load_to_lds(d.g0, d.g1, z, z, 0);
#endif
}

DEV void tdm_store(const TdmDesc& d) {
  v4i z = {0, 0, 0, 0};
#if defined(__clang_major__) && __clang_major__ >= 23
  v8i z8 = {0, 0, 0, 0, 0, 0, 0, 0};
  __builtin_amdgcn_tensor_store_from_lds(d.g0, d.g1, z, z, z8, 0);
#else
  __builtin_amdgcn_tensor_store_from_lds(d.g0, d.g1, z, z, 0);
#endif
}

// ---------------- conversion kernels ----------------
__global__ void k_cvt_f32_f16(const float* __restrict__ in,
                              _Float16* __restrict__ out, int n) {
  int i = blockIdx.x * blockDim.x + threadIdx.x;
  int stride = gridDim.x * blockDim.x;
  for (; i < n; i += stride) out[i] = (_Float16)in[i];
}

// W [R][C] f32 row-major  ->  Wt [C][R] f16 row-major (i.e. W^T)
__global__ __launch_bounds__(256) void k_transpose_f32_f16(
    const float* __restrict__ W, _Float16* __restrict__ Wt, int R, int C) {
  __shared__ __align__(16) _Float16 tile[32][33];
  const int r0 = blockIdx.y * 32, c0 = blockIdx.x * 32;
  const int tx = threadIdx.x & 31, ty = threadIdx.x >> 5;  // 32x8
  for (int i = 0; i < 4; ++i) {
    int r = ty + i * 8;
    tile[r][tx] = (_Float16)W[(size_t)(r0 + r) * C + c0 + tx];
  }
  __syncthreads();
  for (int i = 0; i < 4; ++i) {
    int r = ty + i * 8;
    Wt[(size_t)(c0 + r) * R + r0 + tx] = tile[tx][r];
  }
}

// ---------------- generic WMMA GEMM: C[M,N] = A[M,K] * Bt[N,K]^T ----------------
// Double-buffered: TDM DMAs the next 128x32 tiles into LDS while WMMAs run.
template <bool OUT16>
__global__ __launch_bounds__(256) void k_gemm_f16(
    const _Float16* __restrict__ A, const _Float16* __restrict__ Bt,
    void* __restrict__ Cout, int M, int N, int K) {
  __shared__ __align__(16) _Float16 sA[2][128 * 32];
  __shared__ __align__(16) _Float16 sB[2][128 * 32];
  const int m0 = blockIdx.y * 128;
  const int n0 = blockIdx.x * 128;
  const int t = threadIdx.x;
  const int wid = t >> 5, l = t & 31;
  const int wm = (wid & 3) * 32;   // 4 wave-rows of 32
  const int wn = (wid >> 2) * 64;  // 2 wave-cols of 64
  v8f acc[2][4] = {};
  const int nIter = K / 32;
  if (wid == 0) {  // prologue DMA: tile 0
    tdm_load(make_tdm_2d_f16(lds_u32(sA[0]), A + (size_t)m0 * K, 32, 128, K, M, K));
    tdm_load(make_tdm_2d_f16(lds_u32(sB[0]), Bt + (size_t)n0 * K, 32, 128, K, N, K));
  }
  for (int i = 0; i < nIter; ++i) {
    const int k0 = i * 32;
    if (wid == 0) {
      if (i + 1 < nIter) {  // DMA next tile into other buffer
        tdm_load(make_tdm_2d_f16(lds_u32(sA[(i + 1) & 1]),
                                 A + (size_t)m0 * K + k0 + 32, 32, 128, K, M, K));
        tdm_load(make_tdm_2d_f16(lds_u32(sB[(i + 1) & 1]),
                                 Bt + (size_t)n0 * K + k0 + 32, 32, 128, K, N, K));
        __builtin_amdgcn_s_wait_tensorcnt((short)2);  // tile i complete
      } else {
        __builtin_amdgcn_s_wait_tensorcnt((short)0);
      }
    }
    __syncthreads();  // publish tile i to all waves
    const _Float16* cA = sA[i & 1];
    const _Float16* cB = sB[i & 1];
    v16h bf[4];
    for (int ni = 0; ni < 4; ++ni) bf[ni] = fragB_lds(cB, 32, wn + ni * 16, 0);
    for (int mi = 0; mi < 2; ++mi) {
      v16h af = fragA_lds(cA, 32, wm + mi * 16, 0);
      for (int ni = 0; ni < 4; ++ni) acc[mi][ni] = wmma_f16(af, bf[ni], acc[mi][ni]);
    }
    __syncthreads();  // tile i consumed; its buffer may be re-filled
  }
  const int rlo = (l >> 4) * 8, cc = l & 15;
  for (int mi = 0; mi < 2; ++mi)
    for (int ni = 0; ni < 4; ++ni)
      for (int j = 0; j < 8; ++j) {
        int r = m0 + wm + mi * 16 + j + rlo;
        int c = n0 + wn + ni * 16 + cc;
        if constexpr (OUT16)
          ((_Float16*)Cout)[(size_t)r * N + c] = (_Float16)acc[mi][ni][j];
        else
          ((float*)Cout)[(size_t)r * N + c] = acc[mi][ni][j];
      }
}

// ---------------- bias1: bias[b,h,q,k] = sum_d a_q[b,q,k,d] * K[b,k,h,d] ----------------
__global__ __launch_bounds__(64) void k_bias_aq(const float* __restrict__ a_q,
                                                const _Float16* __restrict__ K16,
                                                float* __restrict__ bias) {
  __shared__ __align__(16) _Float16 sA[64 * 64];
  __shared__ __align__(16) _Float16 sK[16 * 64];
  const int b = blockIdx.x >> 6, kpos = blockIdx.x & 63;
  const int t = threadIdx.x, w = t >> 5, l = t & 31;
  for (int i = 0; i < 64; ++i)
    sA[i * 64 + t] = (_Float16)a_q[(((size_t)(b * 64 + i) * 64 + kpos) << 6) + t];
  const _Float16* kp = K16 + (size_t)(b * 64 + kpos) * 1024;
  for (int i = 0; i < 16; ++i) sK[i * 64 + t] = kp[i * 64 + t];
  __syncthreads();
  v8f acc[2] = {};
  for (int k0 = 0; k0 < 64; k0 += 32) {
    v16h bf = fragB_lds(sK, 64, 0, k0);
    for (int mi = 0; mi < 2; ++mi) {
      v16h af = fragA_lds(sA, 64, (w * 2 + mi) * 16, k0);
      acc[mi] = wmma_f16(af, bf, acc[mi]);
    }
  }
  const int rlo = (l >> 4) * 8, h = l & 15;
  for (int mi = 0; mi < 2; ++mi)
    for (int j = 0; j < 8; ++j) {
      int q = (w * 2 + mi) * 16 + j + rlo;
      bias[(((size_t)(b * 16 + h) * 64 + q) << 6) + kpos] = acc[mi][j];
    }
}

// ---------------- bias2: bias[b,h,q,k] += sum_d Q[b,q,h,d]*a_k[b,q,k,d] ----------------
__global__ __launch_bounds__(64) void k_bias_ak(const float* __restrict__ a_k,
                                                const _Float16* __restrict__ Q16,
                                                float* __restrict__ bias) {
  __shared__ __align__(16) _Float16 sQ[16 * 64];
  __shared__ __align__(16) _Float16 sB[64 * 64];
  const int b = blockIdx.x >> 6, qpos = blockIdx.x & 63;
  const int t = threadIdx.x, w = t >> 5, l = t & 31;
  const _Float16* qp = Q16 + (size_t)(b * 64 + qpos) * 1024;
  for (int i = 0; i < 16; ++i) sQ[i * 64 + t] = qp[i * 64 + t];
  for (int i = 0; i < 64; ++i)
    sB[i * 64 + t] = (_Float16)a_k[(((size_t)(b * 64 + qpos) * 64 + i) << 6) + t];
  __syncthreads();
  v8f acc[2] = {};
  for (int k0 = 0; k0 < 64; k0 += 32) {
    v16h af = fragA_lds(sQ, 64, 0, k0);
    for (int ni = 0; ni < 2; ++ni) {
      v16h bf = fragB_lds(sB, 64, (w * 2 + ni) * 16, k0);
      acc[ni] = wmma_f16(af, bf, acc[ni]);
    }
  }
  const int rlo = (l >> 4) * 8, cc = l & 15;
  for (int ni = 0; ni < 2; ++ni)
    for (int j = 0; j < 8; ++j) {
      int h = j + rlo, k = (w * 2 + ni) * 16 + cc;
      size_t off = (((size_t)(b * 16 + h) * 64 + qpos) << 6) + k;
      bias[off] += acc[ni][j];
    }
}

// ---------------- attention core per (b,h): softmax(QK^T*s + bias) @ V ----------------
__global__ __launch_bounds__(128) void k_attention(
    const _Float16* __restrict__ Q16, const _Float16* __restrict__ K16,
    const _Float16* __restrict__ V16, const float* __restrict__ bias,
    _Float16* __restrict__ attn, _Float16* __restrict__ Pbuf) {
  __shared__ __align__(16) _Float16 sQ[64 * 64];
  __shared__ __align__(16) _Float16 sK[64 * 64];
  __shared__ __align__(16) _Float16 sVt[64 * 64];  // [d][k]
  __shared__ __align__(16) _Float16 sP[64 * 64];
  __shared__ float sS[64 * 64];
  const int bh = blockIdx.x, b = bh >> 4, h = bh & 15;
  const int t = threadIdx.x, w = t >> 5, l = t & 31;
  if (w == 0) {  // TDM: 64x64 head tiles, row stride 1024 elems
    tdm_load(make_tdm_2d_f16(lds_u32(sQ), Q16 + (size_t)(b * 64) * 1024 + h * 64,
                             64, 64, 64, 64, 1024));
    tdm_load(make_tdm_2d_f16(lds_u32(sK), K16 + (size_t)(b * 64) * 1024 + h * 64,
                             64, 64, 64, 64, 1024));
  }
  for (int i = 0; i < 32; ++i) {  // V needs a transpose -> manual staging
    int flat = i * 128 + t;
    int r = flat >> 6, d = flat & 63;
    sVt[d * 64 + r] = V16[(size_t)(b * 64 + r) * 1024 + h * 64 + d];
  }
  if (w == 0) __builtin_amdgcn_s_wait_tensorcnt((short)0);
  __syncthreads();
  const float scale = 0.125f;  // 1/sqrt(64)
  {
    v8f acc[4] = {};
    const int q0 = w * 16;
    for (int k0 = 0; k0 < 64; k0 += 32) {
      v16h af = fragA_lds(sQ, 64, q0, k0);
      for (int ni = 0; ni < 4; ++ni) {
        v16h bf = fragB_lds(sK, 64, ni * 16, k0);
        acc[ni] = wmma_f16(af, bf, acc[ni]);
      }
    }
    const int rlo = (l >> 4) * 8, cc = l & 15;
    for (int ni = 0; ni < 4; ++ni)
      for (int j = 0; j < 8; ++j) {
        int q = q0 + j + rlo, k = ni * 16 + cc;
        sS[q * 64 + k] = acc[ni][j] * scale + bias[(((size_t)bh * 64 + q) << 6) + k];
      }
  }
  __syncthreads();
  if (t < 64) {  // row softmax
    float m = -1e30f;
    for (int k = 0; k < 64; ++k) m = fmaxf(m, sS[t * 64 + k]);
    float s = 0.f;
    for (int k = 0; k < 64; ++k) {
      float e = __expf(sS[t * 64 + k] - m);
      s += e;
      sS[t * 64 + k] = e;
    }
    float inv = 1.f / s;
    for (int k = 0; k < 64; ++k) sP[t * 64 + k] = (_Float16)(sS[t * 64 + k] * inv);
  }
  __syncthreads();
  if (w == 0)  // export P for the a_v pass via TDM store (LDS -> global)
    tdm_store(make_tdm_2d_f16(lds_u32(sP), Pbuf + (size_t)bh * 4096,
                              4096, 1, 4096, 1, 4096));
  {
    v8f acc[4] = {};
    const int q0 = w * 16;
    for (int k0 = 0; k0 < 64; k0 += 32) {
      v16h af = fragA_lds(sP, 64, q0, k0);
      for (int ni = 0; ni < 4; ++ni) {
        v16h bf = fragB_lds(sVt, 64, ni * 16, k0);
        acc[ni] = wmma_f16(af, bf, acc[ni]);
      }
    }
    const int rlo = (l >> 4) * 8, cc = l & 15;
    for (int ni = 0; ni < 4; ++ni)
      for (int j = 0; j < 8; ++j) {
        int q = q0 + j + rlo, d = ni * 16 + cc;
        attn[(size_t)(b * 64 + q) * 1024 + h * 64 + d] = (_Float16)acc[ni][j];
      }
  }
  if (w == 0) __builtin_amdgcn_s_wait_tensorcnt((short)0);  // drain TDM store
}

// ---------------- a_v term: attn[b,q,h,d] += sum_k P[b,h,q,k]*a_v[b,q,k,d] ----------------
__global__ __launch_bounds__(64) void k_av(const float* __restrict__ a_v,
                                           const _Float16* __restrict__ Pbuf,
                                           _Float16* __restrict__ attn) {
  __shared__ __align__(16) _Float16 sP[16 * 64];
  __shared__ __align__(16) _Float16 sAv[64 * 64];  // [d][k]
  const int b = blockIdx.x >> 6, q = blockIdx.x & 63;
  const int t = threadIdx.x, w = t >> 5, l = t & 31;
  if (w == 0)  // TDM: 16 rows of 64, row stride 4096 (= P[b,h,q,:] over h)
    tdm_load(make_tdm_2d_f16(lds_u32(sP),
                             Pbuf + (((size_t)(b * 16) * 64 + q) << 6),
                             64, 16, 64, 16, 4096));
  for (int i = 0; i < 64; ++i)  // read row k=i coalesced (d=t), store transposed
    sAv[t * 64 + i] = (_Float16)a_v[(((size_t)(b * 64 + q) * 64 + i) << 6) + t];
  if (w == 0) __builtin_amdgcn_s_wait_tensorcnt((short)0);
  __syncthreads();
  v8f acc[2] = {};
  for (int k0 = 0; k0 < 64; k0 += 32) {
    v16h af = fragA_lds(sP, 64, 0, k0);
    for (int ni = 0; ni < 2; ++ni) {
      v16h bf = fragB_lds(sAv, 64, (w * 2 + ni) * 16, k0);
      acc[ni] = wmma_f16(af, bf, acc[ni]);
    }
  }
  const int rlo = (l >> 4) * 8, cc = l & 15;
  for (int ni = 0; ni < 2; ++ni)
    for (int j = 0; j < 8; ++j) {
      int hh = j + rlo, d = (w * 2 + ni) * 16 + cc;
      size_t off = (size_t)(b * 64 + q) * 1024 + hh * 64 + d;
      attn[off] = (_Float16)((float)attn[off] + acc[ni][j]);
    }
}

extern "C" void kernel_launch(void* const* d_in, const int* in_sizes, int n_in,
                              void* d_out, int out_size, void* d_ws, size_t ws_size,
                              hipStream_t stream) {
  const float* x   = (const float*)d_in[0];
  const float* a_q = (const float*)d_in[1];
  const float* a_k = (const float*)d_in[2];
  const float* a_v = (const float*)d_in[3];
  const float* Wq  = (const float*)d_in[4];
  const float* Wk  = (const float*)d_in[5];
  const float* Wv  = (const float*)d_in[6];
  const float* Wo  = (const float*)d_in[7];

  char* ws = (char*)d_ws;
  size_t off = 0;
  auto alloc = [&](size_t bytes) {
    char* p = ws + off;
    off += (bytes + 255) & ~(size_t)255;
    return p;
  };
  _Float16* x16  = (_Float16*)alloc(8192ull * 1024 * 2);  // reused as attn16 later
  _Float16* WqT  = (_Float16*)alloc(1024ull * 1024 * 2);
  _Float16* WkT  = (_Float16*)alloc(1024ull * 1024 * 2);
  _Float16* WvT  = (_Float16*)alloc(1024ull * 1024 * 2);
  _Float16* WoT  = (_Float16*)alloc(1024ull * 1024 * 2);
  _Float16* Q16  = (_Float16*)alloc(8192ull * 1024 * 2);
  _Float16* K16  = (_Float16*)alloc(8192ull * 1024 * 2);
  _Float16* V16  = (_Float16*)alloc(8192ull * 1024 * 2);
  _Float16* Pbuf = (_Float16*)alloc(2048ull * 4096 * 2);
  float*    bias = (float*)alloc(2048ull * 4096 * 4);
  _Float16* attn16 = x16;  // x16 dead after QKV projections -> reuse

  // 1) precision conversion (+ weight transpose so B-operand rows are contiguous)
  k_cvt_f32_f16<<<4096, 256, 0, stream>>>(x, x16, 8192 * 1024);
  dim3 tg(32, 32), tb(256);
  k_transpose_f32_f16<<<tg, tb, 0, stream>>>(Wq, WqT, 1024, 1024);
  k_transpose_f32_f16<<<tg, tb, 0, stream>>>(Wk, WkT, 1024, 1024);
  k_transpose_f32_f16<<<tg, tb, 0, stream>>>(Wv, WvT, 1024, 1024);
  k_transpose_f32_f16<<<tg, tb, 0, stream>>>(Wo, WoT, 1024, 1024);

  // 2) Q/K/V projections (WMMA GEMM with TDM double buffering)
  dim3 gg(1024 / 128, 8192 / 128);
  k_gemm_f16<true><<<gg, 256, 0, stream>>>(x16, WqT, Q16, 8192, 1024, 1024);
  k_gemm_f16<true><<<gg, 256, 0, stream>>>(x16, WkT, K16, 8192, 1024, 1024);
  k_gemm_f16<true><<<gg, 256, 0, stream>>>(x16, WvT, V16, 8192, 1024, 1024);

  // 3) smolgen logit biases (each a_* tensor read exactly once, via WMMA)
  k_bias_aq<<<8192, 64, 0, stream>>>(a_q, K16, bias);
  k_bias_ak<<<8192, 64, 0, stream>>>(a_k, Q16, bias);

  // 4) attention core per (b,h)
  k_attention<<<2048, 128, 0, stream>>>(Q16, K16, V16, bias, attn16, Pbuf);

  // 5) per-pair value bias gathered by attention weights
  k_av<<<8192, 64, 0, stream>>>(a_v, Pbuf, attn16);

  // 6) output projection -> fp32 d_out
  k_gemm_f16<false><<<gg, 256, 0, stream>>>(attn16, WoT, (float*)d_out, 8192, 1024, 1024);
}
